// CDK2GNN_81028853006962
// MI455X (gfx1250) — matrix operand order repaired and checked by reading, not compiled, not run on hardware
//
#include <hip/hip_runtime.h>

// ---------------- types ----------------
typedef __bf16 bf16;
typedef __attribute__((ext_vector_type(16))) __bf16 v16bf;
typedef __attribute__((ext_vector_type(8)))  __bf16 v8bf;
typedef __attribute__((ext_vector_type(8)))  float  v8f;

#define NNODES  100000
#define NEDGES  1600000
#define NGRAPHS 2048
// HID = 128 throughout

__device__ __forceinline__ float lrelu01(float v){ return v > 0.f ? v : 0.01f * v; }
// monotone uint key for float atomicMax (init value 0 < key(-inf))
__device__ __forceinline__ unsigned fkey(float f){
    unsigned u = __float_as_uint(f);
    return (u & 0x80000000u) ? ~u : (u | 0x80000000u);
}
__device__ __forceinline__ float funkey(unsigned k){
    unsigned u = (k & 0x80000000u) ? (k & 0x7fffffffu) : ~k;
    return __uint_as_float(u);
}

// A fragment (16-bit, 16x32): lane holds K {koff..koff+7, koff+16..koff+23}
// p already includes row base + kt*32 + koff
__device__ __forceinline__ v16bf load_a_frag(const bf16* p){
    v8bf lo = *(const v8bf*)(p);
    v8bf hi = *(const v8bf*)(p + 16);
    v16bf r;
#pragma unroll
    for (int i = 0; i < 8; ++i){ r[i] = lo[i]; r[i + 8] = hi[i]; }
    return r;
}

// ---------------- weight packing: W[N,K] (out,in) fp32 -> WMMA-B bf16 fragments ----
// B tile (kt,nt): col=nt*16+(lane&15), K = kt*32 + 8h + {0..7, 16..23}, h = lane>>4
__global__ void pack_w(const float* __restrict__ W, bf16* __restrict__ out,
                       int N, int K, int Kp){
    int NT = N >> 4, KT = Kp >> 5;
    int i = blockIdx.x * blockDim.x + threadIdx.x;
    if (i >= KT * NT * 32) return;
    int lane = i & 31; int t = i >> 5;
    int nt = t % NT, kt = t / NT;
    int col = (nt << 4) + (lane & 15);
    int k0  = (kt << 5) + ((lane >> 4) << 3);
    bf16* o = out + ((size_t)(kt * NT + nt) * 32 + lane) * 16;
#pragma unroll
    for (int j = 0; j < 16; ++j){
        int k = k0 + (j & 7) + ((j >> 3) << 4);
        float w = (k < K) ? W[(size_t)col * K + k] : 0.f;
        o[j] = (bf16)w;
    }
}

// fp32 [M,K] -> bf16 [M,Kp] zero-padded
__global__ void cvt_pad(const float* __restrict__ X, bf16* __restrict__ O,
                        long M, int K, int Kp){
    long i = (long)blockIdx.x * blockDim.x + threadIdx.x;
    if (i >= M * (long)Kp) return;
    long m = i / Kp; int k = (int)(i % Kp);
    O[i] = (k < K) ? (bf16)X[m * K + k] : (bf16)0.f;
}

// ---------------- generic GEMM: C = act(A_bf16[M,Kp] @ Bpack + bias) -------------
// block = 256 (8 waves); each wave: 16-row strip, A K-strip held in registers,
// loops over all N/16 col tiles issuing one B-frag load + one WMMA per K step.
template <int KT>
__global__ __launch_bounds__(256)
void gemm_bf16(const bf16* __restrict__ A,
               const bf16* __restrict__ Bp, int N,
               const float* __restrict__ bias,
               float* __restrict__ C, bf16* __restrict__ Cb,
               int M, int act){
    const int Kp = KT * 32;
    int lane = threadIdx.x & 31, wave = threadIdx.x >> 5;
    int rowbase = blockIdx.x * 128 + wave * 16;
    if (rowbase >= M) return;                 // wave-uniform
    int arow = rowbase + (lane & 15);
    if (arow > M - 1) arow = M - 1;
    int koff = (lane >> 4) << 3;
    int NT = N >> 4;
    const bf16* abase = A + (size_t)arow * Kp + koff;
    // preload entire A K-strip into registers (KT*8 VGPRs)
    v16bf av[KT];
#pragma unroll
    for (int kt = 0; kt < KT; ++kt) av[kt] = load_a_frag(abase + kt * 32);
    // speculative prefetch of next row-block of A (gfx1250 global_prefetch)
    {
        int prow = arow + 128; if (prow > M - 1) prow = M - 1;
        __builtin_prefetch(A + (size_t)prow * Kp + koff, 0, 1);
    }
    const bf16* bptr = Bp + (size_t)lane * 16;
    for (int nt = 0; nt < NT; ++nt){
        int col = (nt << 4) + (lane & 15);
        float bv = bias ? bias[col] : 0.f;
        v8f acc = {bv, bv, bv, bv, bv, bv, bv, bv};
#pragma unroll
        for (int kt = 0; kt < KT; ++kt){
            v16bf b = *(const v16bf*)(bptr + ((size_t)(kt * NT + nt) * 32) * 16);
            acc = __builtin_amdgcn_wmma_f32_16x16x32_bf16(false, av[kt], false, b,
                                                          (short)0, acc, false, false);
        }
        int rlo = rowbase + ((lane >> 4) << 3);
#pragma unroll
        for (int i = 0; i < 8; ++i){
            int r = rlo + i;
            if (r < M){
                float v = acc[i];
                if (act == 1) v = lrelu01(v);
                size_t idx = (size_t)r * N + col;
                if (C)  C[idx]  = v;
                if (Cb) Cb[idx] = (bf16)v;
            }
        }
    }
}

// ---------------- GATEConv message GEMM: m = lrelu([x0[src]; eattr] @ lin1^T) ----
// 128 edges per block, A staged (gathered) in LDS [128][160], K-strip in registers
__global__ __launch_bounds__(256)
void gate_msg_gemm(const bf16* __restrict__ x0b, const float* __restrict__ eattr,
                   const int* __restrict__ src, const bf16* __restrict__ Bp,
                   bf16* __restrict__ Mout){
    __shared__ bf16 As[128 * 160];
    int t = threadIdx.x, lane = t & 31, wave = t >> 5;
    long ebase = (long)blockIdx.x * 128;
    {
        int r = t >> 1, half = t & 1;
        long e = ebase + r;
        int s = src[e];
        const bf16* xr = x0b + (size_t)s * 128 + half * 64;
        bf16* dp = As + r * 160 + half * 64;
#pragma unroll
        for (int j = 0; j < 8; ++j) *(v8bf*)(dp + j * 8) = *(const v8bf*)(xr + j * 8);
        if (half){
            bf16* tail = As + r * 160 + 128;
            const float* ea = eattr + e * 8;
#pragma unroll
            for (int c = 0; c < 8; ++c)  tail[c] = (bf16)ea[c];
#pragma unroll
            for (int c = 8; c < 32; ++c) tail[c] = (bf16)0.f;
        }
    }
    __syncthreads();
    int rowl = wave * 16 + (lane & 15);
    int koff = (lane >> 4) << 3;
    const bf16* abase = As + rowl * 160 + koff;
    v16bf av[5];
#pragma unroll
    for (int kt = 0; kt < 5; ++kt) av[kt] = load_a_frag(abase + kt * 32);
    const bf16* bptr = Bp + (size_t)lane * 16;
    for (int nt = 0; nt < 8; ++nt){
        v8f acc = {0.f,0.f,0.f,0.f,0.f,0.f,0.f,0.f};
#pragma unroll
        for (int kt = 0; kt < 5; ++kt){
            v16bf b = *(const v16bf*)(bptr + ((size_t)(kt * 8 + nt) * 32) * 16);
            acc = __builtin_amdgcn_wmma_f32_16x16x32_bf16(false, av[kt], false, b,
                                                          (short)0, acc, false, false);
        }
        int col = (nt << 4) + (lane & 15);
        int rlo = wave * 16 + ((lane >> 4) << 3);
#pragma unroll
        for (int i = 0; i < 8; ++i){
            long e = ebase + rlo + i;
            Mout[(size_t)e * 128 + col] = (bf16)lrelu01(acc[i]);
        }
    }
}

// ---------------- GATEConv aggregation: h[dst] += (m @ lin2^T) * a  (fused) ------
__global__ __launch_bounds__(256)
void gate_agg(const bf16* __restrict__ Mb, const bf16* __restrict__ Bp,
              const float* __restrict__ ealpha, const float* __restrict__ nsum,
              const int* __restrict__ dst, float* __restrict__ H){
    __shared__ float a_l[128];
    __shared__ int   d_l[128];
    int t = threadIdx.x, lane = t & 31, wave = t >> 5;
    long ebase = (long)blockIdx.x * 128;
    if (t < 128){
        long e = ebase + t;
        int d = dst[e];
        d_l[t] = d;
        a_l[t] = ealpha[e] / (nsum[d] + 1e-16f);
    }
    __syncthreads();
    int koff = (lane >> 4) << 3;
    long erow = ebase + wave * 16 + (lane & 15);
    const bf16* abase = Mb + (size_t)erow * 128 + koff;
    v16bf av[4];
#pragma unroll
    for (int kt = 0; kt < 4; ++kt) av[kt] = load_a_frag(abase + kt * 32);
    const bf16* bptr = Bp + (size_t)lane * 16;
    for (int nt = 0; nt < 8; ++nt){
        v8f acc = {0.f,0.f,0.f,0.f,0.f,0.f,0.f,0.f};
#pragma unroll
        for (int kt = 0; kt < 4; ++kt){
            v16bf b = *(const v16bf*)(bptr + ((size_t)(kt * 8 + nt) * 32) * 16);
            acc = __builtin_amdgcn_wmma_f32_16x16x32_bf16(false, av[kt], false, b,
                                                          (short)0, acc, false, false);
        }
        int col = (nt << 4) + (lane & 15);
        int rlo = wave * 16 + ((lane >> 4) << 3);
#pragma unroll
        for (int i = 0; i < 8; ++i){
            int r = rlo + i;
            atomicAdd(&H[(size_t)d_l[r] * 128 + col], acc[i] * a_l[r]);
        }
    }
}

// ---------------- small kernels ----------------
__global__ void rowdot_f32(const float* __restrict__ X, const float* __restrict__ v,
                           float* __restrict__ out, int M){
    int lane = threadIdx.x & 31, wave = threadIdx.x >> 5;
    int row = blockIdx.x * 8 + wave;
    if (row >= M) return;
    const float* xr = X + (size_t)row * 128;
    float s = 0.f;
#pragma unroll
    for (int j = 0; j < 4; ++j){ int c = lane + j * 32; s += xr[c] * v[c]; }
#pragma unroll
    for (int off = 16; off >= 1; off >>= 1) s += __shfl_xor(s, off, 32);
    if (lane == 0) out[row] = s;
}

__global__ void gate_alpha(const bf16* __restrict__ Mb, const float* __restrict__ attl,
                           const float* __restrict__ xr, const int* __restrict__ dst,
                           float* __restrict__ alpha, unsigned* __restrict__ nmax){
    int lane = threadIdx.x & 31, wave = threadIdx.x >> 5;
    long e = (long)blockIdx.x * 8 + wave;
    if (e >= (long)NEDGES) return;
    const bf16* mr = Mb + (size_t)e * 128;
    float s = 0.f;
#pragma unroll
    for (int j = 0; j < 4; ++j){ int c = lane + j * 32; s += (float)mr[c] * attl[c]; }
#pragma unroll
    for (int off = 16; off >= 1; off >>= 1) s += __shfl_xor(s, off, 32);
    if (lane == 0){
        int d = dst[e];
        float a = lrelu01(s + xr[d]);
        alpha[e] = a;
        atomicMax(&nmax[d], fkey(a));
    }
}

__global__ void seg_init(float* __restrict__ H, const float* __restrict__ bias,
                         unsigned* __restrict__ maxk, float* __restrict__ sums, long M){
    long i = (long)blockIdx.x * blockDim.x + threadIdx.x;
    if (i >= M * 128) return;
    int c = (int)(i & 127); long n = i >> 7;
    if (H) H[i] = bias ? bias[c] : 0.f;
    if (c == 0){ if (maxk) maxk[n] = 0u; if (sums) sums[n] = 0.f; }
}

__global__ void seg_exp(const float* __restrict__ alpha, const unsigned* __restrict__ maxk,
                        const int* __restrict__ seg, float* __restrict__ ealpha,
                        float* __restrict__ sums, long E){
    long e = (long)blockIdx.x * blockDim.x + threadIdx.x;
    if (e >= E) return;
    int d = seg[e];
    float ex = __expf(alpha[e] - funkey(maxk[d]));
    ealpha[e] = ex;
    atomicAdd(&sums[d], ex);
}

__global__ void gat_alpha(const float* __restrict__ asrc, const float* __restrict__ adst,
                          const int* __restrict__ src, const int* __restrict__ dst,
                          float* __restrict__ alpha, unsigned* __restrict__ nmax, long E){
    long e = (long)blockIdx.x * blockDim.x + threadIdx.x;
    if (e >= E) return;
    int d = dst[e];
    float a = lrelu01(asrc[src[e]] + adst[d]);
    alpha[e] = a;
    atomicMax(&nmax[d], fkey(a));
}

__global__ void gat_scatter(const float* __restrict__ xp, const float* __restrict__ ealpha,
                            const float* __restrict__ sums, const int* __restrict__ src,
                            const int* __restrict__ dst, float* __restrict__ H, long E){
    long i = (long)blockIdx.x * blockDim.x + threadIdx.x;
    if (i >= E * 32) return;
    long e = i >> 5; int lane = (int)(i & 31);
    int s = src[e], d = dst[e];
    float a = ealpha[e] / (sums[d] + 1e-16f);
    const float* xr = xp + (size_t)s * 128 + lane * 4;
    float* hr = H + (size_t)d * 128 + lane * 4;
#pragma unroll
    for (int j = 0; j < 4; ++j) atomicAdd(&hr[j], xr[j] * a);
}

__global__ void elu_cvt(const float* __restrict__ H, bf16* __restrict__ O, long M){
    long i = (long)blockIdx.x * blockDim.x + threadIdx.x;
    if (i >= M * 128) return;
    float v = H[i];
    v = v > 0.f ? v : (__expf(v) - 1.f);
    O[i] = (bf16)v;
}

__global__ void gru_combine(const float* __restrict__ gi, const float* __restrict__ gh,
                            const float* __restrict__ hprev, float* __restrict__ out,
                            bf16* __restrict__ outb, long M){
    long i = (long)blockIdx.x * blockDim.x + threadIdx.x;
    if (i >= M * 128) return;
    long n = i >> 7; int c = (int)(i & 127);
    long b = n * 384 + c;
    float ir = gi[b], iz = gi[b + 128], inn = gi[b + 256];
    float hr = gh[b], hz = gh[b + 128], hn  = gh[b + 256];
    float r = 1.f / (1.f + __expf(-(ir + hr)));
    float z = 1.f / (1.f + __expf(-(iz + hz)));
    float nn = tanhf(inn + r * hn);
    float v = (1.f - z) * nn + z * hprev[i];
    v = v > 0.f ? v : 0.f;   // relu(gru(...))
    out[i] = v;
    if (outb) outb[i] = (bf16)v;
}

__global__ void readout_scatter(const float* __restrict__ xn, const int* __restrict__ batch,
                                float* __restrict__ gout){
    long i = (long)blockIdx.x * blockDim.x + threadIdx.x;
    if (i >= (long)NNODES * 32) return;
    long n = i >> 5; int lane = (int)(i & 31);
    int g = batch[n];
    const float* xr = xn + (size_t)n * 128 + lane * 4;
    float* orow = gout + (size_t)g * 128 + lane * 4;
#pragma unroll
    for (int j = 0; j < 4; ++j) atomicAdd(&orow[j], xr[j]);
}

__global__ void relu_cvt(float* __restrict__ X, bf16* __restrict__ Xb, long M){
    long i = (long)blockIdx.x * blockDim.x + threadIdx.x;
    if (i >= M * 128) return;
    float v = X[i]; v = v > 0.f ? v : 0.f;
    X[i] = v; Xb[i] = (bf16)v;
}

__global__ void mol_alpha(const float* __restrict__ asrc_n, const float* __restrict__ adst_g,
                          const int* __restrict__ batch, float* __restrict__ nalpha,
                          unsigned* __restrict__ gmax){
    long n = (long)blockIdx.x * blockDim.x + threadIdx.x;
    if (n >= (long)NNODES) return;
    int g = batch[n];
    float a = lrelu01(asrc_n[n] + adst_g[g]);
    nalpha[n] = a;
    atomicMax(&gmax[g], fkey(a));
}

__global__ void mol_scatter(const float* __restrict__ xs, const float* __restrict__ nexp,
                            const float* __restrict__ gsum, const int* __restrict__ batch,
                            float* __restrict__ hg){
    long i = (long)blockIdx.x * blockDim.x + threadIdx.x;
    if (i >= (long)NNODES * 32) return;
    long n = i >> 5; int lane = (int)(i & 31);
    int g = batch[n];
    float w = nexp[n] / (gsum[g] + 1e-16f);
    const float* xr = xs + (size_t)n * 128 + lane * 4;
    float* hr = hg + (size_t)g * 128 + lane * 4;
#pragma unroll
    for (int j = 0; j < 4; ++j) atomicAdd(&hr[j], xr[j] * w);
}

__global__ void final_dot(const float* __restrict__ gout, const float* __restrict__ w2,
                          const float* __restrict__ b2, float* __restrict__ out){
    int lane = threadIdx.x & 31, wave = threadIdx.x >> 5;
    int g = blockIdx.x * 8 + wave;
    if (g >= NGRAPHS) return;
    const float* xr = gout + (size_t)g * 128;
    float s = 0.f;
#pragma unroll
    for (int j = 0; j < 4; ++j){ int c = lane + j * 32; s += xr[c] * w2[c]; }
#pragma unroll
    for (int off = 16; off >= 1; off >>= 1) s += __shfl_xor(s, off, 32);
    if (lane == 0) out[g] = s + b2[0];
}

// ---------------- host ----------------
static inline unsigned cdiv(long a, long b){ return (unsigned)((a + b - 1) / b); }

extern "C" void kernel_launch(void* const* d_in, const int* in_sizes, int n_in,
                              void* d_out, int out_size, void* d_ws, size_t ws_size,
                              hipStream_t stream){
    (void)in_sizes; (void)n_in; (void)out_size; (void)ws_size;
    // inputs (setup_inputs dict order, params flattened in insertion order)
    const float* x      = (const float*)d_in[0];
    const int*   ei     = (const int*)  d_in[1];
    const int*   src    = ei;
    const int*   dstp   = ei + NEDGES;
    const float* eattr  = (const float*)d_in[2];
    const int*   batch  = (const int*)  d_in[3];
    const float* w1     = (const float*)d_in[4];
    const float* b1     = (const float*)d_in[5];
    const float* glin1  = (const float*)d_in[6];
    const float* glin2  = (const float*)d_in[7];
    const float* gattl  = (const float*)d_in[8];
    const float* gattr  = (const float*)d_in[9];
    const float* gbias  = (const float*)d_in[10];
    const float* gru_wi[4] = {(const float*)d_in[11], (const float*)d_in[19],
                              (const float*)d_in[27], (const float*)d_in[36]};
    const float* gru_wh[4] = {(const float*)d_in[12], (const float*)d_in[20],
                              (const float*)d_in[28], (const float*)d_in[37]};
    const float* gru_bi[4] = {(const float*)d_in[13], (const float*)d_in[21],
                              (const float*)d_in[29], (const float*)d_in[38]};
    const float* gru_bh[4] = {(const float*)d_in[14], (const float*)d_in[22],
                              (const float*)d_in[30], (const float*)d_in[39]};
    const float* conv_w[2]    = {(const float*)d_in[15], (const float*)d_in[23]};
    const float* conv_asrc[2] = {(const float*)d_in[16], (const float*)d_in[24]};
    const float* conv_adst[2] = {(const float*)d_in[17], (const float*)d_in[25]};
    const float* conv_bias[2] = {(const float*)d_in[18], (const float*)d_in[26]};
    const float* mol_ws  = (const float*)d_in[31];
    const float* mol_wd  = (const float*)d_in[32];
    const float* matts   = (const float*)d_in[33];
    const float* mattd   = (const float*)d_in[34];
    const float* mbias   = (const float*)d_in[35];
    const float* w2      = (const float*)d_in[40];
    const float* b2      = (const float*)d_in[41];
    float* outp = (float*)d_out;

    // ---- workspace bump allocator ----
    char* base = (char*)d_ws;
    size_t off = 0;
    auto alloc = [&](size_t bytes)->char*{
        char* p = base + off;
        off += (bytes + 255) & ~(size_t)255;
        return p;
    };
    bf16* w1p = (bf16*)alloc(5 * 8 * 32 * 16 * 2);     // Kp=160, N=128
    bf16* g1p = (bf16*)alloc(5 * 8 * 32 * 16 * 2);
    bf16* g2p = (bf16*)alloc(4 * 8 * 32 * 16 * 2);     // 128x128
    bf16 *gip[4], *ghp[4];
    for (int g = 0; g < 4; ++g){
        gip[g] = (bf16*)alloc(4 * 24 * 32 * 16 * 2);   // 128->384
        ghp[g] = (bf16*)alloc(4 * 24 * 32 * 16 * 2);
    }
    bf16* cvp[2] = {(bf16*)alloc(4 * 8 * 32 * 16 * 2), (bf16*)alloc(4 * 8 * 32 * 16 * 2)};
    bf16* wsp  = (bf16*)alloc(4 * 8 * 32 * 16 * 2);
    bf16* wdp  = (bf16*)alloc(4 * 8 * 32 * 16 * 2);

    bf16*  XBF = (bf16*) alloc((size_t)NNODES * 160 * 2);
    float* X0  = (float*)alloc((size_t)NNODES * 128 * 4);
    bf16*  X0B = (bf16*) alloc((size_t)NNODES * 128 * 2);
    float* XN  = (float*)alloc((size_t)NNODES * 128 * 4);
    bf16*  XNB = (bf16*) alloc((size_t)NNODES * 128 * 2);
    float* H   = (float*)alloc((size_t)NNODES * 128 * 4);
    bf16*  EHB = (bf16*) alloc((size_t)NNODES * 128 * 2);
    float* GI  = (float*)alloc((size_t)NNODES * 384 * 4);
    float* GH  = (float*)alloc((size_t)NNODES * 384 * 4);
    char*  MBr = alloc((size_t)NEDGES * 128 * 2);      // big edge buffer, reused below
    bf16*  MB  = (bf16*)MBr;
    float* XP  = (float*)MBr;                          // reused after GATEConv
    float* XS  = (float*)(MBr + (((size_t)NNODES * 128 * 4 + 255) & ~(size_t)255));
    float* ALPHA  = (float*)alloc((size_t)NEDGES * 4);
    float* EALPHA = (float*)alloc((size_t)NEDGES * 4);
    unsigned* NMAX = (unsigned*)alloc((size_t)NNODES * 4);
    float* NSUM = (float*)alloc((size_t)NNODES * 4);
    float* NV1  = (float*)alloc((size_t)NNODES * 4);
    float* NV2  = (float*)alloc((size_t)NNODES * 4);
    float* NV3  = (float*)alloc((size_t)NNODES * 4);
    float* NALPHA = (float*)alloc((size_t)NNODES * 4);
    float* NEXP   = (float*)alloc((size_t)NNODES * 4);
    float* GOUT  = (float*)alloc((size_t)NGRAPHS * 128 * 4);
    bf16*  GOUTB = (bf16*) alloc((size_t)NGRAPHS * 128 * 2);
    float* GTMP  = (float*)alloc((size_t)NGRAPHS * 128 * 4);
    float* GADST = (float*)alloc((size_t)NGRAPHS * 4);
    unsigned* GMAX = (unsigned*)alloc((size_t)NGRAPHS * 4);
    float* GSUM  = (float*)alloc((size_t)NGRAPHS * 4);
    float* HG    = (float*)alloc((size_t)NGRAPHS * 128 * 4);
    bf16*  EHGB  = (bf16*) alloc((size_t)NGRAPHS * 128 * 2);
    float* GGI   = (float*)alloc((size_t)NGRAPHS * 384 * 4);
    float* GGH   = (float*)alloc((size_t)NGRAPHS * 384 * 4);

    dim3 B256(256);
    // ---- pack weights ----
    pack_w<<<cdiv(5*8*32,256), B256, 0, stream>>>(w1,    w1p, 128, 150, 160);
    pack_w<<<cdiv(5*8*32,256), B256, 0, stream>>>(glin1, g1p, 128, 136, 160);
    pack_w<<<cdiv(4*8*32,256), B256, 0, stream>>>(glin2, g2p, 128, 128, 128);
    for (int g = 0; g < 4; ++g){
        pack_w<<<cdiv(4*24*32,256), B256, 0, stream>>>(gru_wi[g], gip[g], 384, 128, 128);
        pack_w<<<cdiv(4*24*32,256), B256, 0, stream>>>(gru_wh[g], ghp[g], 384, 128, 128);
    }
    pack_w<<<cdiv(4*8*32,256), B256, 0, stream>>>(conv_w[0], cvp[0], 128, 128, 128);
    pack_w<<<cdiv(4*8*32,256), B256, 0, stream>>>(conv_w[1], cvp[1], 128, 128, 128);
    pack_w<<<cdiv(4*8*32,256), B256, 0, stream>>>(mol_ws, wsp, 128, 128, 128);
    pack_w<<<cdiv(4*8*32,256), B256, 0, stream>>>(mol_wd, wdp, 128, 128, 128);

    // ---- x0 = lrelu(x @ w1^T + b1) ----
    cvt_pad<<<cdiv((long)NNODES*160,256), B256, 0, stream>>>(x, XBF, NNODES, 150, 160);
    gemm_bf16<5><<<cdiv(NNODES,128), B256, 0, stream>>>(XBF, w1p, 128, b1, X0, X0B, NNODES, 1);
    rowdot_f32<<<cdiv(NNODES,8), B256, 0, stream>>>(X0, gattr, NV1, NNODES);   // x0 @ att_r

    // ---- GATEConv ----
    seg_init<<<cdiv((long)NNODES*128,256), B256, 0, stream>>>(H, gbias, NMAX, NSUM, NNODES);
    gate_msg_gemm<<<NEDGES/128, B256, 0, stream>>>(X0B, eattr, src, g1p, MB);
    gate_alpha<<<cdiv(NEDGES,8), B256, 0, stream>>>(MB, gattl, NV1, dstp, ALPHA, NMAX);
    seg_exp<<<cdiv(NEDGES,256), B256, 0, stream>>>(ALPHA, NMAX, dstp, EALPHA, NSUM, NEDGES);
    gate_agg<<<NEDGES/128, B256, 0, stream>>>(MB, g2p, EALPHA, NSUM, dstp, H);
    elu_cvt<<<cdiv((long)NNODES*128,256), B256, 0, stream>>>(H, EHB, NNODES);
    gemm_bf16<4><<<cdiv(NNODES,128), B256, 0, stream>>>(EHB, gip[0], 384, gru_bi[0], GI, nullptr, NNODES, 0);
    gemm_bf16<4><<<cdiv(NNODES,128), B256, 0, stream>>>(X0B, ghp[0], 384, gru_bh[0], GH, nullptr, NNODES, 0);
    gru_combine<<<cdiv((long)NNODES*128,256), B256, 0, stream>>>(GI, GH, X0, XN, XNB, NNODES);

    // ---- GATConv + GRU layers ----
    for (int l = 0; l < 2; ++l){
        gemm_bf16<4><<<cdiv(NNODES,128), B256, 0, stream>>>(XNB, cvp[l], 128, nullptr, XP, nullptr, NNODES, 0);
        rowdot_f32<<<cdiv(NNODES,8), B256, 0, stream>>>(XP, conv_asrc[l], NV1, NNODES);
        rowdot_f32<<<cdiv(NNODES,8), B256, 0, stream>>>(XP, conv_adst[l], NV2, NNODES);
        seg_init<<<cdiv((long)NNODES*128,256), B256, 0, stream>>>(H, conv_bias[l], NMAX, NSUM, NNODES);
        gat_alpha<<<cdiv(NEDGES,256), B256, 0, stream>>>(NV1, NV2, src, dstp, ALPHA, NMAX, NEDGES);
        seg_exp<<<cdiv(NEDGES,256), B256, 0, stream>>>(ALPHA, NMAX, dstp, EALPHA, NSUM, NEDGES);
        gat_scatter<<<cdiv((long)NEDGES*32,256), B256, 0, stream>>>(XP, EALPHA, NSUM, src, dstp, H, NEDGES);
        elu_cvt<<<cdiv((long)NNODES*128,256), B256, 0, stream>>>(H, EHB, NNODES);
        gemm_bf16<4><<<cdiv(NNODES,128), B256, 0, stream>>>(EHB, gip[l+1], 384, gru_bi[l+1], GI, nullptr, NNODES, 0);
        gemm_bf16<4><<<cdiv(NNODES,128), B256, 0, stream>>>(XNB, ghp[l+1], 384, gru_bh[l+1], GH, nullptr, NNODES, 0);
        gru_combine<<<cdiv((long)NNODES*128,256), B256, 0, stream>>>(GI, GH, XN, XN, XNB, NNODES);
    }

    // ---- readout ----
    seg_init<<<cdiv((long)NGRAPHS*128,256), B256, 0, stream>>>(GOUT, nullptr, nullptr, nullptr, NGRAPHS);
    readout_scatter<<<cdiv((long)NNODES*32,256), B256, 0, stream>>>(XN, batch, GOUT);
    relu_cvt<<<cdiv((long)NGRAPHS*128,256), B256, 0, stream>>>(GOUT, GOUTB, NGRAPHS);
    gemm_bf16<4><<<cdiv(NNODES,128), B256, 0, stream>>>(XNB, wsp, 128, nullptr, XS, nullptr, NNODES, 0);
    rowdot_f32<<<cdiv(NNODES,8), B256, 0, stream>>>(XS, matts, NV3, NNODES);

    for (int t = 0; t < 2; ++t){
        gemm_bf16<4><<<cdiv(NGRAPHS,128), B256, 0, stream>>>(GOUTB, wdp, 128, nullptr, GTMP, nullptr, NGRAPHS, 0);
        rowdot_f32<<<cdiv(NGRAPHS,8), B256, 0, stream>>>(GTMP, mattd, GADST, NGRAPHS);
        seg_init<<<cdiv((long)NGRAPHS*128,256), B256, 0, stream>>>(HG, mbias, GMAX, GSUM, NGRAPHS);
        mol_alpha<<<cdiv(NNODES,256), B256, 0, stream>>>(NV3, GADST, batch, NALPHA, GMAX);
        seg_exp<<<cdiv(NNODES,256), B256, 0, stream>>>(NALPHA, GMAX, batch, NEXP, GSUM, NNODES);
        mol_scatter<<<cdiv((long)NNODES*32,256), B256, 0, stream>>>(XS, NEXP, GSUM, batch, HG);
        elu_cvt<<<cdiv((long)NGRAPHS*128,256), B256, 0, stream>>>(HG, EHGB, NGRAPHS);
        gemm_bf16<4><<<cdiv(NGRAPHS,128), B256, 0, stream>>>(EHGB, gip[3], 384, gru_bi[3], GGI, nullptr, NGRAPHS, 0);
        gemm_bf16<4><<<cdiv(NGRAPHS,128), B256, 0, stream>>>(GOUTB, ghp[3], 384, gru_bh[3], GGH, nullptr, NGRAPHS, 0);
        gru_combine<<<cdiv((long)NGRAPHS*128,256), B256, 0, stream>>>(GGI, GGH, GOUT, GOUT, GOUTB, NGRAPHS);
    }
    final_dot<<<cdiv(NGRAPHS,8), B256, 0, stream>>>(GOUT, w2, b2, outp);
}
